// FutureTrajectoryDecoder_38113539785450
// MI455X (gfx1250) — matrix operand order repaired and checked by reading, not compiled, not run on hardware
//
#include <hip/hip_runtime.h>
#include <hip/hip_bf16.h>

typedef __attribute__((ext_vector_type(16))) __bf16 v16bf;
typedef __attribute__((ext_vector_type(8)))  __bf16 v8bf;
typedef __attribute__((ext_vector_type(8)))  float  v8f;

#define HORIZON 30
#define BTOT    65536
#define INDIM   384
#define LDIH    386
#define DEC     256

#define MT      64      // batch rows per workgroup tile
#define BLOCK   512     // 16 wave32s

// LDS layout (bytes)
//  gi_base: f32, WMMA C-fragment layout [48 ntiles][4 rb][32 lanes][8 f32] = 196608
//  h:       bf16 row-major [64][256+8pad]                                  = 33792
//  mu:      f32 [64][2]                                                    = 512
//  zh:      bf16 row-major [64][384+8pad] (phase-A staging only)           = 50176
#define HP 264
#define ZP 392
#define OFF_GIF 0
#define OFF_H   196608
#define OFF_MU  230400
#define OFF_ZH  230912
#define SMEM_BYTES 281088

// packed bf16 weight regions in workspace (element offsets); 1 fragment = 512 halves = 1KB
#define WS_HH  0                          // W_hh:    48 n-tiles x 8 k-tiles
#define WS_IH  (WS_HH + 48*8*512)         // W_ih[:,:384]: 48 x 12
#define WS_HI  (WS_IH + 48*12*512)        // W_hinit: 16 x 12
#define WS_OUT (WS_HI + 16*12*512)        // [W_mu;W_cov;0-pad] 16x256: 1 x 8
#define WS_TOTAL (WS_OUT + 8*512)

// ---------------------------------------------------------------------------
// Prep: pack all weights into per-lane WMMA B-fragment layout (bf16).
// Fragment (nTile,kTile): lane l<16 -> col nTile*16+l,    K k0..k0+15 ;
//                         lane l>=16 -> col nTile*16+l-16, K k0+16..k0+31.
// ---------------------------------------------------------------------------
__global__ void pack_weights_kernel(const float* __restrict__ Whh,
                                    const float* __restrict__ Wih,
                                    const float* __restrict__ Whinit,
                                    const float* __restrict__ Wmu,
                                    const float* __restrict__ Wcov,
                                    __bf16* __restrict__ ws) {
  int i = blockIdx.x * blockDim.x + threadIdx.x;
  if (i >= WS_TOTAL) return;
  int r    = i & 511;
  int lane = r >> 4;
  int h    = r & 15;
  int khi  = (lane < 16) ? 0 : 16;
  int nsub = lane & 15;
  float val;
  if (i < WS_IH) {                      // W_hh [768][256]
    int f = i >> 9;
    int nt = f >> 3, kt = f & 7;
    val = Whh[(nt*16 + nsub)*DEC + kt*32 + khi + h];
  } else if (i < WS_HI) {               // W_ih [768][386], first 384 K
    int f = (i - WS_IH) >> 9;
    int nt = f / 12, kt = f - nt*12;
    val = Wih[(nt*16 + nsub)*LDIH + kt*32 + khi + h];
  } else if (i < WS_OUT) {              // W_hinit [256][384]
    int f = (i - WS_HI) >> 9;
    int nt = f / 12, kt = f - nt*12;
    val = Whinit[(nt*16 + nsub)*INDIM + kt*32 + khi + h];
  } else {                              // out head: rows 0-1 W_mu, 2-4 W_cov, rest 0
    int kt = (i - WS_OUT) >> 9;
    int k = kt*32 + khi + h;
    val = (nsub < 2) ? Wmu[nsub*DEC + k]
        : ((nsub < 5) ? Wcov[(nsub-2)*DEC + k] : 0.f);
  }
  ws[i] = (__bf16)val;
}

// ---------------------------------------------------------------------------
// Helpers
// ---------------------------------------------------------------------------
union V8F { v8f v; float f[8]; };

__device__ __forceinline__ v16bf load_a(const __bf16* m, int pitch, int row0,
                                        int k0, int lane) {
  // 16-bit A 16x32: lanes 0-15 row M=l, K 0..15; lanes 16-31 row M=l-16, K 16..31
  int rr  = row0 + (lane & 15);
  int off = (lane < 16) ? 0 : 16;
  const __bf16* p = m + rr*pitch + k0 + off;
  union { v16bf v; v8bf h[2]; } u;
  u.h[0] = *(const v8bf*)(p);
  u.h[1] = *(const v8bf*)(p + 8);
  return u.v;
}

__device__ __forceinline__ v8f wmma_bf16(v16bf a, v16bf b, v8f c) {
  return __builtin_amdgcn_wmma_f32_16x16x32_bf16(false, a, false, b, (short)0, c,
                                                 false, false);
}

__device__ __forceinline__ float rcp_fast(float x) {
  return __builtin_amdgcn_rcpf(x);        // raw v_rcp_f32
}
__device__ __forceinline__ float sigmoid_fast(float x) {
  return rcp_fast(1.f + __expf(-x));      // v_exp_f32 + v_rcp_f32
}
__device__ __forceinline__ float tanh_fast(float x) {
#if __has_builtin(__builtin_amdgcn_tanhf)
  return __builtin_amdgcn_tanhf(x);       // v_tanh_f32 (CDNA5 TRANS op)
#else
  return 2.f * rcp_fast(1.f + __expf(-2.f * x)) - 1.f;
#endif
}

// ---------------------------------------------------------------------------
// Persistent GRU rollout: one workgroup owns 64 batch rows for all 30 steps.
// ---------------------------------------------------------------------------
__global__ __launch_bounds__(BLOCK) void gru_rollout_kernel(
    const float* __restrict__ zh,
    const float* __restrict__ Wih,       // for mu rank-2 columns 384,385
    const float* __restrict__ b_hinit, const float* __restrict__ b_ih,
    const float* __restrict__ b_hh,    const float* __restrict__ b_mu,
    const float* __restrict__ b_cov,
    const __bf16* __restrict__ ws,
    float* __restrict__ out) {
  extern __shared__ __align__(32) char smem[];
  v8f*    giF = (v8f*)(smem + OFF_GIF);   // [frag = nt*4+rb][lane] -> v8f
  __bf16* hL  = (__bf16*)(smem + OFF_H);
  float*  muL = (float*)(smem + OFF_MU);
  __bf16* zhL = (__bf16*)(smem + OFF_ZH);

  const int tid  = threadIdx.x;
  const int lane = tid & 31;
  const int wave = tid >> 5;
  const int col  = lane & 15;
  const int hsel = (lane < 16) ? 0 : 8;   // C/D layout: VGPR g -> row g (+8 for hi lanes)
  const int rowBase = blockIdx.x * MT;

  const v16bf* wsHH  = (const v16bf*)(ws + WS_HH);
  const v16bf* wsIH  = (const v16bf*)(ws + WS_IH);
  const v16bf* wsHI  = (const v16bf*)(ws + WS_HI);
  const v16bf* wsOUT = (const v16bf*)(ws + WS_OUT);

  // ---- stage zh_m tile to LDS as bf16; zero mu state ----
  for (int idx = tid; idx < MT*INDIM; idx += BLOCK) {
    int rr = idx / INDIM, cc = idx - rr*INDIM;
    zhL[rr*ZP + cc] = (__bf16)zh[(long)(rowBase + rr)*INDIM + cc];
  }
  if (tid < MT*2) muL[tid] = 0.f;
  __syncthreads();

  // ---- gi_base = zh @ W_ih[:, :384]^T + b_ih -> f32 fragments in LDS ----
  for (int j = 0; j < 3; ++j) {
    int nt = wave + 16*j;                 // n-tile over 48
    v8f acc[4];
    #pragma unroll
    for (int rb = 0; rb < 4; ++rb) acc[rb] = (v8f)(0.f);
    for (int k = 0; k < 12; ++k) {
      v16bf bb = wsIH[(nt*12 + k)*32 + lane];
      #pragma unroll
      for (int rb = 0; rb < 4; ++rb) {
        v16bf aa = load_a(zhL, ZP, rb*16, k*32, lane);
        acc[rb] = wmma_bf16(aa, bb, acc[rb]);
      }
    }
    float bi = b_ih[nt*16 + col];
    #pragma unroll
    for (int rb = 0; rb < 4; ++rb)
      giF[(nt*4 + rb)*32 + lane] = acc[rb] + bi;   // splat add, 2x b128 store
  }

  // ---- h0 = zh @ W_hinit^T + b_hinit -> bf16 LDS [64][256] ----
  {
    V8F acc[4];
    #pragma unroll
    for (int rb = 0; rb < 4; ++rb) acc[rb].v = (v8f)(0.f);
    for (int k = 0; k < 12; ++k) {
      v16bf bb = wsHI[(wave*12 + k)*32 + lane];
      #pragma unroll
      for (int rb = 0; rb < 4; ++rb) {
        v16bf aa = load_a(zhL, ZP, rb*16, k*32, lane);
        acc[rb].v = wmma_bf16(aa, bb, acc[rb].v);
      }
    }
    float bh = b_hinit[wave*16 + col];
    #pragma unroll
    for (int rb = 0; rb < 4; ++rb)
      #pragma unroll
      for (int g = 0; g < 8; ++g) {
        int row = rb*16 + g + hsel;
        hL[row*HP + wave*16 + col] = (__bf16)(acc[rb].f[g] + bh);
      }
  }

  // ---- per-lane loop-invariant constants (this lane's DEC column nG) ----
  const int nG = wave*16 + col;
  const float bhr = b_hh[nG], bhz = b_hh[DEC + nG], bhn = b_hh[2*DEC + nG];
  const float cr0 = Wih[nG*LDIH + 384],          cr1 = Wih[nG*LDIH + 385];
  const float cz0 = Wih[(DEC + nG)*LDIH + 384],  cz1 = Wih[(DEC + nG)*LDIH + 385];
  const float cn0 = Wih[(2*DEC + nG)*LDIH + 384],cn1 = Wih[(2*DEC + nG)*LDIH + 385];
  float obias = 0.f;
  if (col < 2) obias = b_mu[col];
  else if (col < 5) obias = b_cov[col - 2];

  // loop-invariant LDS bases for gate phase
  const v8f* giR0 = giF + ( 0 + wave)*4*32 + lane;   // gate r fragments, +rb*32
  const v8f* giZ0 = giF + (16 + wave)*4*32 + lane;
  const v8f* giN0 = giF + (32 + wave)*4*32 + lane;
  const float2* mu2 = (const float2*)muL;
  __bf16* hCol = hL + hsel*HP + nG;                  // + (rb*16+g)*HP

  __syncthreads();

  // ---- 30-step rollout, h/gi/mu resident in LDS ----
  for (int t = 0; t < HORIZON; ++t) {
    V8F aR[4], aZ[4], aN[4];
    #pragma unroll
    for (int rb = 0; rb < 4; ++rb) {
      aR[rb].v = (v8f)(0.f); aZ[rb].v = (v8f)(0.f); aN[rb].v = (v8f)(0.f);
    }

    for (int k = 0; k < 8; ++k) {        // K = 256 = 8 x 32
      v16bf br = wsHH[((     wave)*8 + k)*32 + lane];
      v16bf bz = wsHH[((16 + wave)*8 + k)*32 + lane];
      v16bf bn = wsHH[((32 + wave)*8 + k)*32 + lane];
      #pragma unroll
      for (int rb = 0; rb < 4; ++rb) {
        v16bf aa = load_a(hL, HP, rb*16, k*32, lane);
        aR[rb].v = wmma_bf16(aa, br, aR[rb].v);
        aZ[rb].v = wmma_bf16(aa, bz, aZ[rb].v);
        aN[rb].v = wmma_bf16(aa, bn, aN[rb].v);
      }
    }
    __syncthreads();                      // all reads of h done before updates

    #pragma unroll
    for (int rb = 0; rb < 4; ++rb) {
      V8F gR, gZ, gN;
      gR.v = giR0[rb*32];                 // 2x ds_load_b128 each
      gZ.v = giZ0[rb*32];
      gN.v = giN0[rb*32];
      #pragma unroll
      for (int g = 0; g < 8; ++g) {
        int row = rb*16 + g + hsel;
        float2 m  = mu2[row];
        float gr = gR.f[g] + m.x*cr0 + m.y*cr1 + aR[rb].f[g] + bhr;
        float gz = gZ.f[g] + m.x*cz0 + m.y*cz1 + aZ[rb].f[g] + bhz;
        float gn = gN.f[g] + m.x*cn0 + m.y*cn1;
        float r_ = sigmoid_fast(gr);
        float z_ = sigmoid_fast(gz);
        float n_ = tanh_fast(gn + r_*(aN[rb].f[g] + bhn));
        float hold = (float)hCol[(rb*16 + g)*HP];
        float hnew = (1.f - z_)*n_ + z_*hold;
        hCol[(rb*16 + g)*HP] = (__bf16)hnew;
      }
    }
    __syncthreads();                      // h_next visible

    if (wave < 4) {                       // output head: [mu;cov] = h_next @ Wout^T
      V8F acc;
      acc.v = (v8f)(0.f);
      for (int k = 0; k < 8; ++k) {
        v16bf bb = wsOUT[k*32 + lane];
        v16bf aa = load_a(hL, HP, wave*16, k*32, lane);
        acc.v = wmma_bf16(aa, bb, acc.v);
      }
      #pragma unroll
      for (int g = 0; g < 8; ++g) {
        int  row  = wave*16 + g + hsel;
        long grow = (long)rowBase + row;
        float v = acc.f[g] + obias;
        long muIdx  = (grow*HORIZON + t)*2;
        long covIdx = (long)BTOT*HORIZON*2 + (grow*HORIZON + t)*4;
        if (col == 0)      { out[muIdx]     = v; muL[row*2]     = v; }
        else if (col == 1) { out[muIdx + 1] = v; muL[row*2 + 1] = v; }
        else if (col == 2) { out[covIdx]     = fminf(fmaxf(v, 0.2f), 1.f); }
        else if (col == 3) { float c = fminf(fmaxf(v, -0.1f), 0.1f);
                             out[covIdx + 1] = c; out[covIdx + 2] = c; }
        else if (col == 4) { out[covIdx + 3] = fminf(fmaxf(v, 0.2f), 1.f); }
      }
    }
    __syncthreads();                      // mu state visible for next step
  }
}

// ---------------------------------------------------------------------------
extern "C" void kernel_launch(void* const* d_in, const int* in_sizes, int n_in,
                              void* d_out, int out_size, void* d_ws, size_t ws_size,
                              hipStream_t stream) {
  const float* zh      = (const float*)d_in[0];
  const float* W_hinit = (const float*)d_in[1];
  const float* b_hinit = (const float*)d_in[2];
  const float* W_ih    = (const float*)d_in[3];
  const float* b_ih    = (const float*)d_in[4];
  const float* W_hh    = (const float*)d_in[5];
  const float* b_hh    = (const float*)d_in[6];
  const float* W_mu    = (const float*)d_in[7];
  const float* b_mu    = (const float*)d_in[8];
  const float* W_cov   = (const float*)d_in[9];
  const float* b_cov   = (const float*)d_in[10];
  float* out = (float*)d_out;
  __bf16* ws = (__bf16*)d_ws;

  (void)in_sizes; (void)n_in; (void)out_size; (void)ws_size;

  pack_weights_kernel<<<(WS_TOTAL + 255) / 256, 256, 0, stream>>>(
      W_hh, W_ih, W_hinit, W_mu, W_cov, ws);

  (void)hipFuncSetAttribute((const void*)gru_rollout_kernel,
                            hipFuncAttributeMaxDynamicSharedMemorySize,
                            SMEM_BYTES);
  gru_rollout_kernel<<<BTOT / MT, BLOCK, SMEM_BYTES, stream>>>(
      zh, W_ih, b_hinit, b_ih, b_hh, b_mu, b_cov, ws, out);
}
